// PdistMixBert_49632642073334
// MI455X (gfx1250) — compile-verified, matrix operand order checked by reading, not run on hardware
//
#include <hip/hip_runtime.h>

// ---------------- model constants ----------------
#define Bc   16
#define Lc   512
#define Dc   768
#define Hc   12
#define KDc  64
#define VDc  64
#define Fc   3072
#define NLc  12
#define NCc  30
#define DQ   768            // H*KD
#define NTOK (Bc*Lc)        // 8192
#define BCH  4              // attention batch chunk
#define CHTOK 2048          // FFN token chunk

// ---------------- GEMM tile ----------------
#define BM 256              // 8 waves x 32 rows
#define BN 64
#define BK 32
#define LDS_ROW 40          // halves per LDS row: BK + 8 pad (80B stride, 16B aligned)

typedef __attribute__((ext_vector_type(16))) __bf16 v16bf;
typedef __attribute__((ext_vector_type(8)))  __bf16 v8bf;
typedef __attribute__((ext_vector_type(8)))  float  v8f;

__device__ __forceinline__ __bf16 f2bf(float f) {
  union { float f; unsigned u; } a; a.f = f;
  unsigned r = a.u + 0x7FFFu + ((a.u >> 16) & 1u);      // round-nearest-even
  union { unsigned short s; __bf16 h; } o; o.s = (unsigned short)(r >> 16);
  return o.h;
}

__device__ __forceinline__ float geluf(float x) {
  return 0.5f * x * (1.0f + erff(x * 0.70710678118654752440f));
}

// CDNA5 async memory->LDS copy (16B per lane), tracked by ASYNCcnt
__device__ __forceinline__ void async_copy16(const void* gptr, void* lptr) {
  const unsigned lds = (unsigned)(unsigned long long)lptr;   // low 32b = LDS offset
  asm volatile("global_load_async_to_lds_b128 %0, %1, off"
               :: "v"(lds), "v"((unsigned long long)gptr)
               : "memory");
}
__device__ __forceinline__ void wait_async0() {
  asm volatile("s_wait_asynccnt 0x0" ::: "memory");
}

__device__ __forceinline__ float blockReduceSum(float v, float* red) {
  const int lane = threadIdx.x & 31, wave = threadIdx.x >> 5;
  #pragma unroll
  for (int o = 16; o > 0; o >>= 1) v += __shfl_xor(v, o, 32);
  if (lane == 0) red[wave] = v;
  __syncthreads();
  const int nw = blockDim.x >> 5;
  float r = red[0];
  for (int w = 1; w < nw; ++w) r += red[w];
  __syncthreads();
  return r;
}

__device__ __forceinline__ float blockReduceMax(float v, float* red) {
  const int lane = threadIdx.x & 31, wave = threadIdx.x >> 5;
  #pragma unroll
  for (int o = 16; o > 0; o >>= 1) v = fmaxf(v, __shfl_xor(v, o, 32));
  if (lane == 0) red[wave] = v;
  __syncthreads();
  const int nw = blockDim.x >> 5;
  float r = red[0];
  for (int w = 1; w < nw; ++w) r = fmaxf(r, red[w]);
  __syncthreads();
  return r;
}

// ============================================================
// Batched bf16 GEMM: C = act(scale * A@B^T + bias)
//   A: bf16 [M][K] (k-contiguous, stride lda)
//   B: bf16 [N][K] (k-contiguous, stride ldb)
// All M/N/K exact multiples of BM/BN/BK. Double-buffered LDS,
// async global->LDS copies overlapped with WMMA.
// Wave tile 32x64: B-fragments reused across 2 A-rows (8 WMMA / 12 ds-loads).
// OUT: 0 = f32 C, 1 = bf16 C, 2 = bf16 V-transpose scatter ([b][h][d][j])
// ============================================================
template<int ACT, int OUT>
__global__ __launch_bounds__(256)
void gemm_bf16(const __bf16* __restrict__ A, const __bf16* __restrict__ Bm,
               const float* __restrict__ bias, void* __restrict__ Cv,
               int M, int N, int K, int lda, int ldb, int ldc,
               long long sAb, long long sAh, long long sBb, long long sBh,
               long long sCb, long long sCh,
               int Hdim, const int* __restrict__ bRemap, float scale)
{
  __shared__ __bf16 sA[2][BM * LDS_ROW];
  __shared__ __bf16 sB[2][BN * LDS_ROW];

  const int z  = blockIdx.z;
  const int bb = z / Hdim;
  const int hh = z - bb * Hdim;
  const int rb = bRemap ? bRemap[bb] : bb;
  A  += (long long)bb * sAb + (long long)hh * sAh;
  Bm += (long long)rb * sBb + (long long)hh * sBh;
  const long long cbase = (long long)bb * sCb + (long long)hh * sCh;

  const int m0 = blockIdx.y * BM;
  const int n0 = blockIdx.x * BN;
  const int tid  = threadIdx.x;
  const int wave = tid >> 5;
  const int lane = tid & 31;
  const int l16  = lane & 15;
  const int kb   = (lane >> 4) << 3;   // 0 or 8: ISA 16-bit operand lane->K mapping

  // ---- loop-invariant per-thread copy addresses (4 A chunks + 1 B chunk) ----
  const int tr = tid >> 2;             // 0..63
  const int tc = (tid & 3) * 8;        // 0/8/16/24
  const __bf16* gA[4];
  gA[0] = A + (long long)(m0 + tr) * lda + tc;
  gA[1] = gA[0] + 64ll * lda;
  gA[2] = gA[0] + 128ll * lda;
  gA[3] = gA[0] + 192ll * lda;
  const __bf16* gB1 = Bm + (long long)(n0 + tr) * ldb + tc;
  __bf16* lA[2][4]; __bf16* lB1[2];
  #pragma unroll
  for (int u = 0; u < 2; ++u) {
    #pragma unroll
    for (int p = 0; p < 4; ++p) lA[u][p] = &sA[u][(tr + 64 * p) * LDS_ROW + tc];
    lB1[u] = &sB[u][tr * LDS_ROW + tc];
  }

  v8f acc[2][4] = {};
  const int KT = K / BK;

  // prologue: stage tile 0 into buffer 0
  #pragma unroll
  for (int p = 0; p < 4; ++p) async_copy16(gA[p], lA[0][p]);
  async_copy16(gB1, lB1[0]);
  wait_async0();
  __syncthreads();

  for (int kt = 0; kt < KT; ++kt) {
    const int cur = kt & 1;
    if (kt + 1 < KT) {                  // overlap next-tile copy with WMMA
      const int k1 = (kt + 1) * BK;
      #pragma unroll
      for (int p = 0; p < 4; ++p) async_copy16(gA[p] + k1, lA[cur ^ 1][p]);
      async_copy16(gB1 + k1, lB1[cur ^ 1]);
    }

    // A fragments for this wave's two 16-row strips
    v16bf afrag[2];
    #pragma unroll
    for (int mi = 0; mi < 2; ++mi) {
      const int arow = ((wave << 5) + (mi << 4) + l16) * LDS_ROW + kb;
      const v8bf alo = *(const v8bf*)&sA[cur][arow];
      const v8bf ahi = *(const v8bf*)&sA[cur][arow + 16];
      afrag[mi] = __builtin_shufflevector(alo, ahi,
          0,1,2,3,4,5,6,7,8,9,10,11,12,13,14,15);
    }
    #pragma unroll
    for (int nt = 0; nt < 4; ++nt) {
      const int brow = ((nt << 4) + l16) * LDS_ROW + kb;
      const v8bf blo = *(const v8bf*)&sB[cur][brow];
      const v8bf bhi = *(const v8bf*)&sB[cur][brow + 16];
      const v16bf bfrag = __builtin_shufflevector(blo, bhi,
          0,1,2,3,4,5,6,7,8,9,10,11,12,13,14,15);
      #pragma unroll
      for (int mi = 0; mi < 2; ++mi) {
        acc[mi][nt] = __builtin_amdgcn_wmma_f32_16x16x32_bf16(
            false, afrag[mi], false, bfrag, (short)0, acc[mi][nt], false, false);
      }
    }

    if (kt + 1 < KT) {
      wait_async0();                    // this wave's async copies landed
      __syncthreads();                  // all waves' copies landed
    }
  }

  // ---- epilogue: C/D layout vgpr j -> M = j + 8*(lane>=16), lane%16 -> N ----
  #pragma unroll
  for (int mi = 0; mi < 2; ++mi) {
    const int mbase = m0 + (wave << 5) + (mi << 4) + ((lane >> 4) << 3);
    #pragma unroll
    for (int nt = 0; nt < 4; ++nt) {
      const int gn = n0 + (nt << 4) + l16;
      const float bvv = bias ? bias[gn] : 0.0f;
      #pragma unroll
      for (int j = 0; j < 8; ++j) {
        const int gm = mbase + j;
        float v = acc[mi][nt][j] * scale + bvv;
        if (ACT == 1) v = geluf(v);
        if (OUT == 0) {
          ((float*)Cv)[cbase + (long long)gm * ldc + gn] = v;
        } else if (OUT == 1) {
          ((__bf16*)Cv)[cbase + (long long)gm * ldc + gn] = f2bf(v);
        } else {
          // V-transpose scatter: gm = token (b*L+j2), gn = h*64+d -> [b][h][d][j2]
          const int bidx = gm >> 9;          // /Lc
          const int jj   = gm & (Lc - 1);
          const int hh2  = gn >> 6;
          const int dd   = gn & 63;
          ((__bf16*)Cv)[(((long long)bidx * Hc + hh2) * VDc + dd) * Lc + jj] = f2bf(v);
        }
      }
    }
  }
}

// ============================================================
// Weight transpose + convert: W f32 [K][N] -> Wt bf16 [N][K]
// ============================================================
__global__ __launch_bounds__(256)
void transpose_to_bf16(const float* __restrict__ W, __bf16* __restrict__ Wt,
                       int K, int N)
{
  __shared__ float tile[32][33];
  const int n0 = blockIdx.x * 32, k0 = blockIdx.y * 32;
  const int tx = threadIdx.x, ty = threadIdx.y;
  #pragma unroll
  for (int r = ty; r < 32; r += 8)
    tile[r][tx] = W[(long long)(k0 + r) * N + n0 + tx];
  __syncthreads();
  #pragma unroll
  for (int r = ty; r < 32; r += 8)
    Wt[(long long)(n0 + r) * K + k0 + tx] = f2bf(tile[tx][r]);
}

// ============================================================
// Embedding + LayerNorm: writes f32 h and bf16 copy
// ============================================================
__global__ __launch_bounds__(256)
void embed_ln_kernel(const int* __restrict__ ids,
                     const float* __restrict__ we, const float* __restrict__ pe,
                     const float* __restrict__ te,
                     const float* __restrict__ w, const float* __restrict__ b,
                     float* __restrict__ h, __bf16* __restrict__ hb)
{
  const int t = blockIdx.x;
  const int l = t % Lc;
  const int id = ids[t];
  __shared__ float red[8];
  float x[4]; int cnt = 0;
  float s = 0.0f;
  for (int d = threadIdx.x; d < Dc; d += 256) {
    float v = we[(long long)id * Dc + d] + pe[(long long)l * Dc + d] + te[d];
    x[cnt++] = v; s += v;
  }
  const float mu = blockReduceSum(s, red) / (float)Dc;
  float s2 = 0.0f;
  for (int u = 0; u < cnt; ++u) { float c = x[u] - mu; s2 += c * c; }
  const float var = blockReduceSum(s2, red) / (float)Dc;
  const float inv = rsqrtf(var + 1e-12f);
  cnt = 0;
  for (int d = threadIdx.x; d < Dc; d += 256) {
    float v = (x[cnt++] - mu) * inv * w[d] + b[d];
    h[(long long)t * Dc + d] = v;
    hb[(long long)t * Dc + d] = f2bf(v);
  }
}

// ============================================================
// h = LN(h + y); writes f32 h and bf16 copy
// ============================================================
__global__ __launch_bounds__(256)
void add_ln_kernel(float* __restrict__ h, const float* __restrict__ y,
                   const float* __restrict__ w, const float* __restrict__ b,
                   __bf16* __restrict__ hb)
{
  const int t = blockIdx.x;
  __shared__ float red[8];
  float x[4]; int cnt = 0;
  float s = 0.0f;
  for (int d = threadIdx.x; d < Dc; d += 256) {
    float v = h[(long long)t * Dc + d] + y[(long long)t * Dc + d];
    x[cnt++] = v; s += v;
  }
  const float mu = blockReduceSum(s, red) / (float)Dc;
  float s2 = 0.0f;
  for (int u = 0; u < cnt; ++u) { float c = x[u] - mu; s2 += c * c; }
  const float var = blockReduceSum(s2, red) / (float)Dc;
  const float inv = rsqrtf(var + 1e-12f);
  cnt = 0;
  for (int d = threadIdx.x; d < Dc; d += 256) {
    float v = (x[cnt++] - mu) * inv * w[d] + b[d];
    h[(long long)t * Dc + d] = v;
    hb[(long long)t * Dc + d] = f2bf(v);
  }
}

// ============================================================
// Masked softmax over rows of f32 [z][L][L]; writes f32 in-place + bf16 probs
// ============================================================
__global__ __launch_bounds__(256)
void softmax_rows(float* __restrict__ S, __bf16* __restrict__ P,
                  const float* __restrict__ mask, const int* __restrict__ remap,
                  int Hdim, int L)
{
  const int i = blockIdx.x;
  const int z = blockIdx.y;
  const int bb = z / Hdim;
  const int mb = remap ? remap[bb] : bb;
  float* row = S + ((long long)z * L + i) * L;
  __bf16* prow = P + ((long long)z * L + i) * L;
  const float* mrow = mask + (long long)mb * L;
  __shared__ float red[8];

  float mv[2], mk[2];
  float mx = -3.402823466e+38f;
  #pragma unroll
  for (int u = 0; u < 2; ++u) {
    int j = threadIdx.x + u * 256;
    float m = (j < L) ? mrow[j] : 0.0f;
    float v = (j < L) ? row[j] * m : 0.0f;
    mv[u] = v; mk[u] = m;
    mx = fmaxf(mx, v);
  }
  mx = blockReduceMax(mx, red);
  float s = 0.0f;
  #pragma unroll
  for (int u = 0; u < 2; ++u) {
    float e = __expf(mv[u] - mx) * mk[u];
    mv[u] = e; s += e;
  }
  s = blockReduceSum(s, red);
  s = s + ((s == 0.0f) ? 1.0f : 0.0f);
  const float invs = 1.0f / s;
  #pragma unroll
  for (int u = 0; u < 2; ++u) {
    int j = threadIdx.x + u * 256;
    if (j < L) {
      float e = mv[u] * invs;
      row[j] = e;
      prow[j] = f2bf(e);
    }
  }
}

// ============================================================
// cross_sim = max over heads of attention probs (per batch-chunk)
// ============================================================
__global__ __launch_bounds__(256)
void head_max_kernel(const float* __restrict__ S, float* __restrict__ sim, int total)
{
  const int gi = blockIdx.x * 256 + threadIdx.x;
  if (gi >= total) return;
  const long long LL = (long long)Lc * Lc;
  const int b = gi / (int)LL;
  const int r = gi - b * (int)LL;
  long long base = (long long)b * Hc * LL + r;
  float m = S[base];
  #pragma unroll
  for (int h = 1; h < Hc; ++h) m = fmaxf(m, S[base + (long long)h * LL]);
  sim[gi] = m;
}

// ============================================================
// masked argmax over j (mask = same positional category), tie -> larger j
// ============================================================
__global__ __launch_bounds__(256)
void masked_argmax_kernel(const float* __restrict__ sim, int* __restrict__ pos, int c0)
{
  const int bb = blockIdx.y;
  const int i  = blockIdx.x;
  const int cat_i = (i == 0) ? 0 : ((i == Lc - 1) ? 2 : 1);
  const float* row = sim + ((long long)bb * Lc + i) * Lc;

  float bestv = -3.402823466e+38f;
  int bestj = -1;
  for (int j = threadIdx.x; j < Lc; j += 256) {
    const int cat_j = (j == 0) ? 0 : ((j == Lc - 1) ? 2 : 1);
    if (cat_j == cat_i) {
      float v = row[j];
      if (v > bestv || (v == bestv && j > bestj)) { bestv = v; bestj = j; }
    }
  }
  __shared__ float rv[256];
  __shared__ int   rj[256];
  rv[threadIdx.x] = bestv; rj[threadIdx.x] = bestj;
  __syncthreads();
  for (int s = 128; s > 0; s >>= 1) {
    if (threadIdx.x < s) {
      float v2 = rv[threadIdx.x + s]; int j2 = rj[threadIdx.x + s];
      if (v2 > rv[threadIdx.x] || (v2 == rv[threadIdx.x] && j2 > rj[threadIdx.x])) {
        rv[threadIdx.x] = v2; rj[threadIdx.x] = j2;
      }
    }
    __syncthreads();
  }
  if (threadIdx.x == 0) pos[(long long)(c0 + bb) * Lc + i] = (rj[0] < 0) ? 0 : rj[0];
}

// ============================================================
// h2[b,i,:] = h[mix[b], pos[b,i], :]   (f32 gather)
// ============================================================
__global__ __launch_bounds__(256)
void gather_rows(const float* __restrict__ h, const int* __restrict__ mix,
                 const int* __restrict__ pos, float* __restrict__ h2)
{
  const long long gi = (long long)blockIdx.x * 256 + threadIdx.x;
  const long long total = (long long)NTOK * Dc;
  if (gi >= total) return;
  const int tok = (int)(gi / Dc);
  const int d   = (int)(gi - (long long)tok * Dc);
  const int b   = tok / Lc;
  h2[gi] = h[((long long)mix[b] * Lc + pos[tok]) * Dc + d];
}

// ============================================================
// h = alpha*h + (1-alpha)*h2; refresh bf16 copy
// ============================================================
__global__ __launch_bounds__(256)
void blend_kernel(float* __restrict__ h, const float* __restrict__ h2,
                  const float* __restrict__ alpha, long long n,
                  __bf16* __restrict__ hb)
{
  const long long gi = (long long)blockIdx.x * 256 + threadIdx.x;
  if (gi >= n) return;
  const float a = alpha[0];
  const float v = a * h[gi] + (1.0f - a) * h2[gi];
  h[gi] = v;
  hb[gi] = f2bf(v);
}

// ============================================================
// pooled[b,:] = mean over L of h[b,:,:]
// ============================================================
__global__ __launch_bounds__(256)
void pool_mean(const float* __restrict__ h, float* __restrict__ pooled)
{
  const int b = blockIdx.x;
  for (int d = threadIdx.x; d < Dc; d += 256) {
    float s = 0.0f;
    for (int l = 0; l < Lc; ++l) s += h[((long long)b * Lc + l) * Dc + d];
    pooled[(long long)b * Dc + d] = s / (float)Lc;
  }
}

__global__ __launch_bounds__(128)
void head1_kernel(const float* __restrict__ pooled, const float* __restrict__ w,
                  const float* __restrict__ bias, float* __restrict__ t)
{
  const int b = blockIdx.x;
  const int j = threadIdx.x;
  float s = bias[j];
  for (int d = 0; d < Dc; ++d) s += pooled[(long long)b * Dc + d] * w[(long long)d * 128 + j];
  t[b * 128 + j] = tanhf(s);
}

__global__ __launch_bounds__(32)
void head2_kernel(const float* __restrict__ t, const float* __restrict__ w,
                  const float* __restrict__ bias, float* __restrict__ out)
{
  const int b = blockIdx.x;
  const int c = threadIdx.x;
  if (c >= NCc) return;
  float s = bias[c];
  for (int j = 0; j < 128; ++j) s += t[b * 128 + j] * w[j * NCc + c];
  out[b * NCc + c] = s;
}

// ============================================================
// Host orchestration
// ============================================================
extern "C" void kernel_launch(void* const* d_in, const int* in_sizes, int n_in,
                              void* d_out, int out_size, void* d_ws, size_t ws_size,
                              hipStream_t stream)
{
  (void)in_sizes; (void)n_in; (void)out_size; (void)ws_size;

  const int*   ids   = (const int*)  d_in[0];
  const float* amask = (const float*)d_in[1];
  const int*   mixp  = (const int*)  d_in[2];
  const float* alpha = (const float*)d_in[6];
  const float* wemb  = (const float*)d_in[7];
  const float* pemb  = (const float*)d_in[8];
  const float* temb  = (const float*)d_in[9];
  const float* elnw  = (const float*)d_in[10];
  const float* elnb  = (const float*)d_in[11];
  const float* Wq_   = (const float*)d_in[12];
  const float* bq_   = (const float*)d_in[13];
  const float* Wk_   = (const float*)d_in[14];
  const float* bk_   = (const float*)d_in[15];
  const float* Wv_   = (const float*)d_in[16];
  const float* bv_   = (const float*)d_in[17];
  const float* Wo_   = (const float*)d_in[18];
  const float* bo_   = (const float*)d_in[19];
  const float* l1w   = (const float*)d_in[20];
  const float* l1b   = (const float*)d_in[21];
  const float* Wf1_  = (const float*)d_in[22];
  const float* bf1_  = (const float*)d_in[23];
  const float* Wf2_  = (const float*)d_in[24];
  const float* bf2_  = (const float*)d_in[25];
  const float* l2w   = (const float*)d_in[26];
  const float* l2b   = (const float*)d_in[27];
  const float* c1w   = (const float*)d_in[28];
  const float* c1b   = (const float*)d_in[29];
  const float* c2w   = (const float*)d_in[30];
  const float* c2b   = (const float*)d_in[31];

  float* ws = (float*)d_ws;
  size_t off = 0;
  auto alloc = [&](size_t fl) { float* p = ws + off; off += (fl + 3) & ~(size_t)3; return p; };

  const size_t TOKD = (size_t)NTOK * Dc;                 // 6,291,456 elements
  float*  pHf  = alloc(TOKD);                            // f32 residual stream
  float*  pG   = alloc(TOKD);                            // f32 GEMM out / h2 gather
  float*  pS   = alloc((size_t)BCH * Hc * Lc * Lc);      // f32 attention logits
  float*  pSim = alloc((size_t)BCH * Lc * Lc);           // f32 cross-sim
  __bf16* pHb  = (__bf16*)alloc(TOKD / 2);               // bf16 h copy
  __bf16* pQb  = (__bf16*)alloc(TOKD / 2);
  __bf16* pKb  = (__bf16*)alloc(TOKD / 2);
  __bf16* pVt  = (__bf16*)alloc(TOKD / 2);               // V transposed [b][h][d][j]
  __bf16* pO   = (__bf16*)alloc(TOKD / 2);               // attention output
  __bf16* pP   = (__bf16*)alloc((size_t)BCH * Hc * Lc * Lc / 2);  // bf16 probs
  __bf16* pFb  = (__bf16*)alloc((size_t)CHTOK * Fc / 2); // bf16 gelu outputs
  __bf16* pWt  = (__bf16*)alloc((size_t)7077888 / 2);    // per-layer bf16 W^T
  int*    pPos = (int*)alloc(NTOK);
  float*  pPool = alloc((size_t)Bc * Dc);
  float*  pT    = alloc((size_t)Bc * 128);

  __bf16* WtQ  = pWt;
  __bf16* WtK  = pWt + (size_t)1 * 589824;
  __bf16* WtV  = pWt + (size_t)2 * 589824;
  __bf16* WtO  = pWt + (size_t)3 * 589824;
  __bf16* WtF1 = pWt + (size_t)4 * 589824;               // [3072][768]
  __bf16* WtF2 = WtF1 + (size_t)2359296;                 // [768][3072]

  const float scl = 0.125f;                              // 1/sqrt(KD)
  const long long sLQ = (long long)Lc * DQ;              // per-batch Q/K stride
  const long long sHS = (long long)Hc * Lc * Lc;         // per-batch score stride
  const long long sLL = (long long)Lc * Lc;              // per-head score stride
  const long long sVb = (long long)Hc * VDc * Lc;        // per-batch V^T stride
  const long long sVh = (long long)VDc * Lc;             // per-head V^T stride
  const dim3 TB(32, 8);

  embed_ln_kernel<<<NTOK, 256, 0, stream>>>(ids, wemb, pemb, temb, elnw, elnb, pHf, pHb);

  for (int i = 0; i < NLc; ++i) {
    const float* wq  = Wq_  + (size_t)i * Dc * DQ;  const float* bqi  = bq_  + (size_t)i * DQ;
    const float* wk  = Wk_  + (size_t)i * Dc * DQ;  const float* bki  = bk_  + (size_t)i * DQ;
    const float* wv  = Wv_  + (size_t)i * Dc * DQ;  const float* bvi  = bv_  + (size_t)i * DQ;
    const float* wo  = Wo_  + (size_t)i * DQ * Dc;  const float* boi  = bo_  + (size_t)i * Dc;
    const float* wf1 = Wf1_ + (size_t)i * Dc * Fc;  const float* bf1i = bf1_ + (size_t)i * Fc;
    const float* wf2 = Wf2_ + (size_t)i * Fc * Dc;  const float* bf2i = bf2_ + (size_t)i * Dc;

    // ---- per-layer weight transpose + bf16 convert ----
    transpose_to_bf16<<<dim3(DQ/32, Dc/32), TB, 0, stream>>>(wq,  WtQ,  Dc, DQ);
    transpose_to_bf16<<<dim3(DQ/32, Dc/32), TB, 0, stream>>>(wk,  WtK,  Dc, DQ);
    transpose_to_bf16<<<dim3(DQ/32, Dc/32), TB, 0, stream>>>(wv,  WtV,  Dc, DQ);
    transpose_to_bf16<<<dim3(Dc/32, DQ/32), TB, 0, stream>>>(wo,  WtO,  DQ, Dc);
    transpose_to_bf16<<<dim3(Fc/32, Dc/32), TB, 0, stream>>>(wf1, WtF1, Dc, Fc);
    transpose_to_bf16<<<dim3(Dc/32, Fc/32), TB, 0, stream>>>(wf2, WtF2, Fc, Dc);

    if (i == 0) {
      // -------- mixup: cross attention on pre-mix h --------
      dim3 gp(DQ / BN, NTOK / BM, 1);
      gemm_bf16<0, 1><<<gp, 256, 0, stream>>>(pHb, WtQ, bqi, pQb,
          NTOK, DQ, Dc, Dc, Dc, DQ, 0, 0, 0, 0, 0, 0, 1, nullptr, 1.0f);
      gemm_bf16<0, 1><<<gp, 256, 0, stream>>>(pHb, WtK, bki, pKb,
          NTOK, DQ, Dc, Dc, Dc, DQ, 0, 0, 0, 0, 0, 0, 1, nullptr, 1.0f);

      for (int c0 = 0; c0 < Bc; c0 += BCH) {
        dim3 gs(Lc / BN, Lc / BM, BCH * Hc);
        gemm_bf16<0, 0><<<gs, 256, 0, stream>>>(pQb + (size_t)c0 * sLQ, pKb, nullptr, pS,
            Lc, Lc, KDc, DQ, DQ, Lc,
            sLQ, KDc, sLQ, KDc, sHS, sLL, Hc, mixp + c0, scl);
        softmax_rows<<<dim3(Lc, BCH * Hc), 256, 0, stream>>>(pS, pP, amask, mixp + c0, Hc, Lc);
        const int totalSim = BCH * Lc * Lc;
        head_max_kernel<<<(totalSim + 255) / 256, 256, 0, stream>>>(pS, pSim, totalSim);
        masked_argmax_kernel<<<dim3(Lc, BCH), 256, 0, stream>>>(pSim, pPos, c0);
      }
      gather_rows<<<(int)((TOKD + 255) / 256), 256, 0, stream>>>(pHf, mixp, pPos, pG);
      blend_kernel<<<(int)((TOKD + 255) / 256), 256, 0, stream>>>(pHf, pG, alpha,
          (long long)TOKD, pHb);
    }

    // -------- Q, K, V projections (bf16 out; V written transposed) --------
    {
      dim3 g(DQ / BN, NTOK / BM, 1);
      gemm_bf16<0, 1><<<g, 256, 0, stream>>>(pHb, WtQ, bqi, pQb,
          NTOK, DQ, Dc, Dc, Dc, DQ, 0, 0, 0, 0, 0, 0, 1, nullptr, 1.0f);
      gemm_bf16<0, 1><<<g, 256, 0, stream>>>(pHb, WtK, bki, pKb,
          NTOK, DQ, Dc, Dc, Dc, DQ, 0, 0, 0, 0, 0, 0, 1, nullptr, 1.0f);
      gemm_bf16<0, 2><<<g, 256, 0, stream>>>(pHb, WtV, bvi, pVt,
          NTOK, DQ, Dc, Dc, Dc, DQ, 0, 0, 0, 0, 0, 0, 1, nullptr, 1.0f);
    }

    // -------- attention, chunked over batch --------
    for (int c0 = 0; c0 < Bc; c0 += BCH) {
      dim3 gs(Lc / BN, Lc / BM, BCH * Hc);
      gemm_bf16<0, 0><<<gs, 256, 0, stream>>>(pQb + (size_t)c0 * sLQ,
          pKb + (size_t)c0 * sLQ, nullptr, pS,
          Lc, Lc, KDc, DQ, DQ, Lc,
          sLQ, KDc, sLQ, KDc, sHS, sLL, Hc, nullptr, scl);
      softmax_rows<<<dim3(Lc, BCH * Hc), 256, 0, stream>>>(pS, pP,
          amask + (size_t)c0 * Lc, nullptr, Hc, Lc);
      dim3 ga(VDc / BN, Lc / BM, BCH * Hc);
      gemm_bf16<0, 1><<<ga, 256, 0, stream>>>(pP, pVt + (size_t)c0 * sVb, nullptr,
          pO + (size_t)c0 * sLQ,
          Lc, VDc, Lc, Lc, Lc, DQ,
          sHS, sLL, sVb, sVh, sLQ, VDc, Hc, nullptr, 1.0f);
    }

    // -------- output projection (f32) + residual LN --------
    {
      dim3 g(Dc / BN, NTOK / BM, 1);
      gemm_bf16<0, 0><<<g, 256, 0, stream>>>(pO, WtO, boi, pG,
          NTOK, Dc, DQ, DQ, DQ, Dc, 0, 0, 0, 0, 0, 0, 1, nullptr, 1.0f);
    }
    add_ln_kernel<<<NTOK, 256, 0, stream>>>(pHf, pG,
        l1w + (size_t)i * Dc, l1b + (size_t)i * Dc, pHb);

    // -------- FFN, chunked over tokens --------
    for (int c0 = 0; c0 < NTOK; c0 += CHTOK) {
      dim3 g1(Fc / BN, CHTOK / BM, 1);
      gemm_bf16<1, 1><<<g1, 256, 0, stream>>>(pHb + (size_t)c0 * Dc, WtF1, bf1i, pFb,
          CHTOK, Fc, Dc, Dc, Dc, Fc, 0, 0, 0, 0, 0, 0, 1, nullptr, 1.0f);
      dim3 g2(Dc / BN, CHTOK / BM, 1);
      gemm_bf16<0, 0><<<g2, 256, 0, stream>>>(pFb, WtF2, bf2i, pG + (size_t)c0 * Dc,
          CHTOK, Dc, Fc, Fc, Fc, Dc, 0, 0, 0, 0, 0, 0, 1, nullptr, 1.0f);
    }
    add_ln_kernel<<<NTOK, 256, 0, stream>>>(pHf, pG,
        l2w + (size_t)i * Dc, l2b + (size_t)i * Dc, pHb);
  }

  // -------- pooled classifier head --------
  pool_mean<<<Bc, 256, 0, stream>>>(pHf, pPool);
  head1_kernel<<<Bc, 128, 0, stream>>>(pPool, c1w, c1b, pT);
  head2_kernel<<<Bc, 32, 0, stream>>>(pT, c2w, c2b, (float*)d_out);
}
